// GNN_42305427865769
// MI455X (gfx1250) — compile-verified, matrix operand order checked by reading, not run on hardware
//
#include <hip/hip_runtime.h>
#include <hip/hip_bf16.h>

#define LL 3
#define BB 8192
#define KK 1024
#define DD 512

#define BETA_C   1e-4f
#define GAMMA_C  1.0f
#define LAMBD_C  0.1f
#define OT_EPS_C 0.1f
#define OT_ITER_C 50
#define TEMP_NCE_C 0.07f

typedef __bf16 bf16_t;
typedef __attribute__((ext_vector_type(8)))  bf16_t v8bf;
typedef __attribute__((ext_vector_type(16))) bf16_t v16bf;
typedef __attribute__((ext_vector_type(8)))  float  v8f;
typedef int v4i_g __attribute__((vector_size(16)));

// LDS tile: 128 rows x 32 bf16, padded to 40 elem stride (80B) -> conflict-free b128 reads
#define LDS_STRIDE 40
#define TILE_ELEMS (128 * LDS_STRIDE)
#define KSTEPS (DD / 32)

// ---------- CDNA5 async global->LDS path (with portable fallback) ----------

#if __has_builtin(__builtin_amdgcn_global_load_async_to_lds_b128)
#define HAVE_ASYNC_LDS 1
#else
#define HAVE_ASYNC_LDS 0
#endif

__device__ __forceinline__ void async_cp16(const bf16_t* g, bf16_t* l) {
#if HAVE_ASYNC_LDS
  __builtin_amdgcn_global_load_async_to_lds_b128(
      (__attribute__((address_space(1))) v4i_g*)(g),
      (__attribute__((address_space(3))) v4i_g*)(l), 0, 0);
#else
  *(v8bf*)l = *(const v8bf*)g;   // sync fallback: global_load_b128 + ds_store_b128
#endif
}

__device__ __forceinline__ void wait_async0() {
#if HAVE_ASYNC_LDS
#if __has_builtin(__builtin_amdgcn_s_wait_asynccnt)
  __builtin_amdgcn_s_wait_asynccnt(0);
#else
  asm volatile("s_wait_asynccnt 0" ::: "memory");
#endif
#endif
}

// ---------- small device helpers ----------

__device__ __forceinline__ unsigned enc_f(float f) {
  unsigned b = __float_as_uint(f);
  return (b & 0x80000000u) ? ~b : (b | 0x80000000u);
}
__device__ __forceinline__ float dec_f(unsigned k) {
  unsigned b = (k & 0x80000000u) ? (k ^ 0x80000000u) : ~k;
  return __uint_as_float(b);
}
__device__ __forceinline__ void atomicMinFloat(float* addr, float v) {
  unsigned old = *(volatile unsigned*)addr;
  while (__uint_as_float(old) > v) {
    unsigned prev = atomicCAS((unsigned*)addr, old, __float_as_uint(v));
    if (prev == old) break;
    old = prev;
  }
}

// ---------- init kernels ----------

__global__ void fill_u64_kernel(unsigned long long* p, unsigned long long v, int n) {
  int i = blockIdx.x * 256 + threadIdx.x;
  if (i < n) p[i] = v;
}
__global__ void fill_f32_kernel(float* p, float v, int n) {
  int i = blockIdx.x * 256 + threadIdx.x;
  if (i < n) p[i] = v;
}

// ---------- e = mu + exp(ls)*eps, KL accumulation ----------

__global__ void build_e_kernel(const float* __restrict__ mu, const float* __restrict__ ls,
                               const float* __restrict__ eps, float* __restrict__ e,
                               float* __restrict__ total) {
  __shared__ float sm[256];
  int t = threadIdx.x;
  int i = blockIdx.x * 256 + t;                 // i < K*D
  float m = mu[i], l = ls[i], p = eps[i];
  float s = __expf(l);
  e[i] = m + s * p;
  float term = m * m + s * s - 1.0f - 2.0f * l; // mu^2 + exp(2ls) - 1 - 2ls
  sm[t] = term; __syncthreads();
  #pragma unroll
  for (int st = 128; st > 0; st >>= 1) { if (t < st) sm[t] += sm[t + st]; __syncthreads(); }
  if (t == 0) atomicAdd(total, (0.5f * BETA_C / (float)KK) * sm[0]);
}

// ---------- per-row: sq-norm, bf16 copy, normalized bf16 copy ----------

__global__ void rows_prep_kernel(const float* __restrict__ rows, float* __restrict__ sqn,
                                 bf16_t* __restrict__ bf, bf16_t* __restrict__ nbf) {
  __shared__ float sm[256];
  int r = blockIdx.x, t = threadIdx.x;
  const float* p = rows + (size_t)r * DD;
  float v0 = p[t], v1 = p[t + 256];
  sm[t] = v0 * v0 + v1 * v1; __syncthreads();
  #pragma unroll
  for (int st = 128; st > 0; st >>= 1) { if (t < st) sm[t] += sm[t + st]; __syncthreads(); }
  float ss = sm[0];
  if (t == 0 && sqn) sqn[r] = ss;
  if (bf)  { bf[(size_t)r * DD + t] = (bf16_t)v0; bf[(size_t)r * DD + t + 256] = (bf16_t)v1; }
  if (nbf) {
    float inv = rsqrtf(ss);
    nbf[(size_t)r * DD + t] = (bf16_t)(v0 * inv);
    nbf[(size_t)r * DD + t + 256] = (bf16_t)(v1 * inv);
  }
}

// ---------- WMMA tile engine with double-buffered async LDS staging ----------
// Block tile 128(M) x 128(N), k-chunk 32. 8 waves as 2(M) x 4(N); each wave: 4x2 16x16 tiles.
// Fragment layout (16x32 bf16): lane L -> row L&15, kbase=(L>>4)*8 ; elems 0..7 = K[kbase..+7],
// elems 8..15 = K[kbase+16..+23]. B operand (N x K view of E rows) uses the identical layout.

__device__ __forceinline__ void stage_tile(const bf16_t* __restrict__ g, int rowBase,
                                           bf16_t* lbuf, int tid, int k0) {
  int lr = tid >> 1;                 // 0..127 tile row
  int lc = (tid & 1) * 16;           // element offset in row (two 16B chunks per thread)
  const bf16_t* gs = g + (size_t)(rowBase + lr) * DD + k0 + lc;
  bf16_t* ld = lbuf + lr * LDS_STRIDE + lc;
  if (k0 + 64 < DD)                  // pull next-next chunk toward L2 (global_prefetch_b8)
    __builtin_prefetch((const void*)(gs + 64), 0, 1);
  async_cp16(gs, ld);
  async_cp16(gs + 8, ld + 8);
}

__device__ __forceinline__ v16bf lds_frag(const bf16_t* buf, int row, int hi) {
  const bf16_t* p = buf + row * LDS_STRIDE + hi * 8;
  v8bf lo = *(const v8bf*)p;
  v8bf h2 = *(const v8bf*)(p + 16);
  return __builtin_shufflevector(lo, h2, 0,1,2,3,4,5,6,7,8,9,10,11,12,13,14,15);
}

__device__ __forceinline__ void gemm_tiles(const bf16_t* __restrict__ A, const bf16_t* __restrict__ Bm,
                                           int ArowBase, int BrowBase, int tid,
                                           v8f (&acc)[4][2], bf16_t* lsA, bf16_t* lsB) {
  int lane = tid & 31, wave = tid >> 5;
  int wm = wave >> 2, wn = wave & 3;
  int rlo = lane & 15, hi = lane >> 4;

  // prologue: fill buffer 0
  stage_tile(A,  ArowBase, lsA, tid, 0);
  stage_tile(Bm, BrowBase, lsB, tid, 0);

  for (int ks = 0; ks < KSTEPS; ++ks) {
    int p = ks & 1;
    wait_async0();          // my async writes into buf[p] have landed
    __syncthreads();        // everyone's writes visible; prev iter's reads of buf[1-p] done
    if (ks + 1 < KSTEPS) {  // issue async copies for the next chunk into the other buffer
      stage_tile(A,  ArowBase, lsA + (1 - p) * TILE_ELEMS, tid, (ks + 1) * 32);
      stage_tile(Bm, BrowBase, lsB + (1 - p) * TILE_ELEMS, tid, (ks + 1) * 32);
    }
    const bf16_t* Ab = lsA + p * TILE_ELEMS;
    const bf16_t* Bb = lsB + p * TILE_ELEMS;
    v16bf b0 = lds_frag(Bb, wn * 32 + rlo,      hi);
    v16bf b1 = lds_frag(Bb, wn * 32 + 16 + rlo, hi);
    #pragma unroll
    for (int i = 0; i < 4; i++) {
      v16bf a = lds_frag(Ab, wm * 64 + i * 16 + rlo, hi);
      acc[i][0] = __builtin_amdgcn_wmma_f32_16x16x32_bf16(false, a, false, b0, (short)0, acc[i][0], false, false);
      acc[i][1] = __builtin_amdgcn_wmma_f32_16x16x32_bf16(false, a, false, b1, (short)0, acc[i][1], false, false);
    }
  }
}

// ---------- fused distance GEMM + row argmin (packed u64 atomicMin) + column min ----------

__global__ __launch_bounds__(256)
void dist_wmma_kernel(const bf16_t* __restrict__ X, const bf16_t* __restrict__ E,
                      const float* __restrict__ xn, const float* __restrict__ en,
                      unsigned long long* __restrict__ rowkey, float* __restrict__ colmin) {
  __shared__ __attribute__((aligned(16))) bf16_t lsA[2 * TILE_ELEMS];
  __shared__ __attribute__((aligned(16))) bf16_t lsB[2 * TILE_ELEMS];
  int tid = threadIdx.x;
  int lane = tid & 31, wave = tid >> 5;
  int wm = wave >> 2, wn = wave & 3;
  int rowBase = blockIdx.y * 128 + wm * 64;
  int colBase = blockIdx.x * 128 + wn * 32;
  v8f acc[4][2] = {};
  gemm_tiles(X, E, blockIdx.y * 128, blockIdx.x * 128, tid, acc, lsA, lsB);

  int rlo = lane & 15, hi = lane >> 4;
  float en0 = en[colBase + rlo];
  float en1 = en[colBase + 16 + rlo];
  float cmin0 = 3.4e38f, cmin1 = 3.4e38f;
  #pragma unroll
  for (int i = 0; i < 4; i++) {
    unsigned long long rk[8];
    #pragma unroll
    for (int j = 0; j < 8; j++) {
      int row = rowBase + i * 16 + j + 8 * hi;
      float x2 = xn[row];
      float d0 = x2 + en0 - 2.0f * acc[i][0][j];
      float d1 = x2 + en1 - 2.0f * acc[i][1][j];
      cmin0 = fminf(cmin0, d0); cmin1 = fminf(cmin1, d1);
      unsigned long long k0v = (((unsigned long long)enc_f(d0)) << 32) | (unsigned)(colBase + rlo);
      unsigned long long k1v = (((unsigned long long)enc_f(d1)) << 32) | (unsigned)(colBase + 16 + rlo);
      rk[j] = k0v < k1v ? k0v : k1v;
    }
    #pragma unroll
    for (int m = 1; m < 16; m <<= 1) {
      #pragma unroll
      for (int j = 0; j < 8; j++) {
        unsigned long long o = __shfl_xor(rk[j], m, 32);
        rk[j] = (o < rk[j]) ? o : rk[j];
      }
    }
    if (rlo == 0) {
      #pragma unroll
      for (int j = 0; j < 8; j++)
        atomicMin(&rowkey[rowBase + i * 16 + j + 8 * hi], rk[j]);
    }
  }
  cmin0 = fminf(cmin0, __shfl_xor(cmin0, 16, 32));
  cmin1 = fminf(cmin1, __shfl_xor(cmin1, 16, 32));
  if (hi == 0) {
    atomicMinFloat(&colmin[colBase + rlo], cmin0);
    atomicMinFloat(&colmin[colBase + 16 + rlo], cmin1);
  }
}

// ---------- generic NT GEMM with two epilogues (Gram store / sqdist + exp kernel) ----------

__global__ __launch_bounds__(256)
void gemm_nt_kernel(const bf16_t* __restrict__ A, const bf16_t* __restrict__ Bm,
                    const float* __restrict__ an, const float* __restrict__ bn,
                    float* __restrict__ out0, float* __restrict__ out1, int mode, int N) {
  __shared__ __attribute__((aligned(16))) bf16_t lsA[2 * TILE_ELEMS];
  __shared__ __attribute__((aligned(16))) bf16_t lsB[2 * TILE_ELEMS];
  int tid = threadIdx.x;
  int lane = tid & 31, wave = tid >> 5;
  int wm = wave >> 2, wn = wave & 3;
  int rowBase = blockIdx.y * 128 + wm * 64;
  int colBase = blockIdx.x * 128 + wn * 32;
  v8f acc[4][2] = {};
  gemm_tiles(A, Bm, blockIdx.y * 128, blockIdx.x * 128, tid, acc, lsA, lsB);

  int rlo = lane & 15, hi = lane >> 4;
  #pragma unroll
  for (int i = 0; i < 4; i++) {
    #pragma unroll
    for (int tc = 0; tc < 2; tc++) {
      int col = colBase + tc * 16 + rlo;
      #pragma unroll
      for (int j = 0; j < 8; j++) {
        int row = rowBase + i * 16 + j + 8 * hi;
        float s = acc[i][tc][j];
        if (mode == 0) {
          out0[(size_t)row * N + col] = s;
        } else {
          float c = an[row] + bn[col] - 2.0f * s;
          out0[(size_t)row * N + col] = c;
          out1[(size_t)row * N + col] = __expf(-c * (1.0f / OT_EPS_C));
        }
      }
    }
  }
}

// ---------- epilogues for VQ losses / idx / counts ----------

__global__ void finalize_rows_kernel(const unsigned long long* __restrict__ rowkey,
                                     int* __restrict__ out_idx, int l,
                                     float* __restrict__ cnt_l, float* __restrict__ total) {
  __shared__ float sm[256];
  int t = threadIdx.x;
  int b = blockIdx.x * 256 + t;
  unsigned long long k = rowkey[b];
  int idx = (int)(unsigned)(k & 0xFFFFFFFFull);
  float d = dec_f((unsigned)(k >> 32));
  out_idx[(size_t)b * LL + l] = idx;
  atomicAdd(&cnt_l[idx], 1.0f);
  sm[t] = d; __syncthreads();
  #pragma unroll
  for (int st = 128; st > 0; st >>= 1) { if (t < st) sm[t] += sm[t + st]; __syncthreads(); }
  if (t == 0) atomicAdd(total, 2.0f * sm[0] / ((float)BB * (float)DD));
}

__global__ void finalize_cols_kernel(const float* __restrict__ colmin, float* __restrict__ total) {
  __shared__ float sm[256];
  int t = threadIdx.x;
  int k = blockIdx.x * 256 + t;
  sm[t] = colmin[k]; __syncthreads();
  #pragma unroll
  for (int st = 128; st > 0; st >>= 1) { if (t < st) sm[t] += sm[t + st]; __syncthreads(); }
  if (t == 0) atomicAdd(total, 2.0f * sm[0] / ((float)KK * (float)DD));
}

__global__ void gather_q_kernel(const int* __restrict__ out_idx, const float* __restrict__ e,
                                float* __restrict__ out_q, int l) {
  int b = blockIdx.x, t = threadIdx.x;
  int idx = out_idx[(size_t)b * LL + l];
  const float* src = e + (size_t)idx * DD;
  float* dst = out_q + (size_t)b * LL * DD + (size_t)l * DD;
  dst[t] = src[t];
  dst[t + 256] = src[t + 256];
}

// ---------- InfoNCE via K x K Gram ----------

__global__ void lse_kernel(const float* __restrict__ G, const float* __restrict__ cntp,
                           float* __restrict__ lse) {
  __shared__ float sm[256];
  int c = blockIdx.x, t = threadIdx.x;
  const float* g = G + (size_t)c * KK;
  float m = -3.4e38f;
  for (int k = t; k < KK; k += 256) m = fmaxf(m, g[k] * (1.0f / TEMP_NCE_C));
  sm[t] = m; __syncthreads();
  #pragma unroll
  for (int st = 128; st > 0; st >>= 1) { if (t < st) sm[t] = fmaxf(sm[t], sm[t + st]); __syncthreads(); }
  m = sm[0]; __syncthreads();
  float s = 0.0f;
  for (int k = t; k < KK; k += 256) s += cntp[k] * __expf(g[k] * (1.0f / TEMP_NCE_C) - m);
  sm[t] = s; __syncthreads();
  #pragma unroll
  for (int st = 128; st > 0; st >>= 1) { if (t < st) sm[t] += sm[t + st]; __syncthreads(); }
  if (t == 0) lse[c] = m + __logf(sm[0]);
}

__global__ void nce_accum_kernel(const int* __restrict__ out_idx, const float* __restrict__ G,
                                 const float* __restrict__ lse, float* __restrict__ total, int l) {
  __shared__ float sm[256];
  int t = threadIdx.x;
  int b = blockIdx.x * 256 + t;
  int c  = out_idx[(size_t)b * LL + l];
  int kp = out_idx[(size_t)b * LL + l - 1];
  float term = G[(size_t)c * KK + kp] * (1.0f / TEMP_NCE_C) - lse[c];
  sm[t] = term; __syncthreads();
  #pragma unroll
  for (int st = 128; st > 0; st >>= 1) { if (t < st) sm[t] += sm[t + st]; __syncthreads(); }
  if (t == 0) atomicAdd(total, -LAMBD_C * sm[0] / (float)BB);
}

// ---------- Sinkhorn: single 1024-thread block, u/v in LDS ----------

__global__ __launch_bounds__(1024)
void sinkhorn_kernel(const float* __restrict__ Kmat, const float* __restrict__ cost,
                     const float* __restrict__ cprev, const float* __restrict__ ccur,
                     float* __restrict__ total) {
  __shared__ float u[KK], v[KK], red[1024];
  int t = threadIdx.x;
  float mu = cprev[t] / (float)BB + 1e-8f;
  float nu = ccur[t]  / (float)BB + 1e-8f;
  u[t] = 1.0f; v[t] = 1.0f; __syncthreads();
  for (int it = 0; it < OT_ITER_C; ++it) {
    const float* kr = Kmat + (size_t)t * KK;
    float s = 0.0f;
    for (int k = 0; k < KK; k++) s += kr[k] * v[k];
    float ut = mu / s;
    __syncthreads(); u[t] = ut; __syncthreads();
    float s2 = 0.0f;
    for (int k = 0; k < KK; k++) s2 += Kmat[(size_t)k * KK + t] * u[k];
    float vt = nu / s2;
    __syncthreads(); v[t] = vt; __syncthreads();
  }
  const float* kr = Kmat + (size_t)t * KK;
  const float* cr = cost + (size_t)t * KK;
  float s = 0.0f;
  for (int k = 0; k < KK; k++) s += kr[k] * v[k] * cr[k];
  s *= u[t];
  red[t] = s; __syncthreads();
  #pragma unroll
  for (int st = 512; st > 0; st >>= 1) { if (t < st) red[t] += red[t + st]; __syncthreads(); }
  if (t == 0) atomicAdd(total, GAMMA_C * red[0]);
}

__global__ void write_total_kernel(const float* __restrict__ acc, float* __restrict__ out) {
  out[0] = acc[0];
}

// ---------- host ----------

extern "C" void kernel_launch(void* const* d_in, const int* in_sizes, int n_in,
                              void* d_out, int out_size, void* d_ws, size_t ws_size,
                              hipStream_t stream) {
  (void)in_sizes; (void)n_in; (void)out_size; (void)ws_size;
  const float* latents = (const float*)d_in[0]; // (L,B,D)
  const float* mus     = (const float*)d_in[1]; // (L,K,D)
  const float* logsig  = (const float*)d_in[2]; // (L,K,D)
  const float* eps     = (const float*)d_in[3]; // (L,K,D)

  int*   out_idx   = (int*)d_out;                                   // (B,L)
  float* out_q     = (float*)d_out + (size_t)BB * LL;               // (B,L,D)
  float* out_total = (float*)d_out + (size_t)BB * LL + (size_t)BB * LL * DD;

  char* w = (char*)d_ws;
  size_t off = 0;
  auto carve = [&](size_t bytes) -> void* {
    off = (off + 255) & ~(size_t)255;
    void* p = w + off;
    off += bytes;
    return p;
  };
  float*  e_f32   = (float*) carve((size_t)LL * KK * DD * 4);
  bf16_t* e_bf    = (bf16_t*)carve((size_t)LL * KK * DD * 2);
  bf16_t* en_bf   = (bf16_t*)carve((size_t)LL * KK * DD * 2);
  bf16_t* mu_bf   = (bf16_t*)carve((size_t)LL * KK * DD * 2);
  bf16_t* x_bf    = (bf16_t*)carve((size_t)BB * DD * 2);
  float*  xn2     = (float*) carve((size_t)BB * 4);
  float*  en2     = (float*) carve((size_t)LL * KK * 4);
  float*  mun2    = (float*) carve((size_t)LL * KK * 4);
  unsigned long long* rowkey = (unsigned long long*)carve((size_t)BB * 8);
  float*  colmin  = (float*) carve((size_t)KK * 4);
  float*  cnt     = (float*) carve((size_t)LL * KK * 4);
  float*  Gmat    = (float*) carve((size_t)KK * KK * 4);
  float*  costm   = (float*) carve((size_t)KK * KK * 4);
  float*  Kker    = (float*) carve((size_t)KK * KK * 4);
  float*  lse     = (float*) carve((size_t)KK * 4);
  float*  total   = (float*) carve(256);

  // zero accumulators
  fill_f32_kernel<<<1, 256, 0, stream>>>(total, 0.0f, 1);
  fill_f32_kernel<<<(LL * KK + 255) / 256, 256, 0, stream>>>(cnt, 0.0f, LL * KK);

  for (int l = 0; l < LL; ++l) {
    const float* mu_l = mus    + (size_t)l * KK * DD;
    const float* ls_l = logsig + (size_t)l * KK * DD;
    const float* ep_l = eps    + (size_t)l * KK * DD;
    const float* x_l  = latents + (size_t)l * BB * DD;
    float*  e_l   = e_f32 + (size_t)l * KK * DD;
    bf16_t* eb_l  = e_bf  + (size_t)l * KK * DD;
    bf16_t* enb_l = en_bf + (size_t)l * KK * DD;
    bf16_t* mb_l  = mu_bf + (size_t)l * KK * DD;

    // e = mu + exp(ls)*eps ; KL -> total
    build_e_kernel<<<(KK * DD) / 256, 256, 0, stream>>>(mu_l, ls_l, ep_l, e_l, total);
    // per-row prep: e (norms + bf16 + normalized bf16), x (norms + bf16), mu (norms + bf16)
    rows_prep_kernel<<<KK, 256, 0, stream>>>(e_l, en2 + (size_t)l * KK, eb_l, enb_l);
    rows_prep_kernel<<<BB, 256, 0, stream>>>(x_l, xn2, x_bf, nullptr);
    rows_prep_kernel<<<KK, 256, 0, stream>>>(mu_l, mun2 + (size_t)l * KK, mb_l, nullptr);

    // distance GEMM + argmin
    fill_u64_kernel<<<BB / 256, 256, 0, stream>>>(rowkey, ~0ull, BB);
    fill_f32_kernel<<<KK / 256, 256, 0, stream>>>(colmin, 3.4e38f, KK);
    dist_wmma_kernel<<<dim3(KK / 128, BB / 128), 256, 0, stream>>>(
        x_bf, eb_l, xn2, en2 + (size_t)l * KK, rowkey, colmin);

    finalize_rows_kernel<<<BB / 256, 256, 0, stream>>>(rowkey, out_idx, l, cnt + (size_t)l * KK, total);
    finalize_cols_kernel<<<KK / 256, 256, 0, stream>>>(colmin, total);
    gather_q_kernel<<<BB, 256, 0, stream>>>(out_idx, e_l, out_q, l);

    if (l > 0) {
      // InfoNCE: G = En_l @ En_{l-1}^T  (K x K)
      gemm_nt_kernel<<<dim3(KK / 128, KK / 128), 256, 0, stream>>>(
          enb_l, en_bf + (size_t)(l - 1) * KK * DD, nullptr, nullptr, Gmat, nullptr, 0, KK);
      lse_kernel<<<KK, 256, 0, stream>>>(Gmat, cnt + (size_t)(l - 1) * KK, lse);
      nce_accum_kernel<<<BB / 256, 256, 0, stream>>>(out_idx, Gmat, lse, total, l);

      // Sinkhorn: cost = sqdist(mus[l-1], mus[l]); Kmat = exp(-cost/eps)
      gemm_nt_kernel<<<dim3(KK / 128, KK / 128), 256, 0, stream>>>(
          mu_bf + (size_t)(l - 1) * KK * DD, mb_l,
          mun2 + (size_t)(l - 1) * KK, mun2 + (size_t)l * KK, costm, Kker, 1, KK);
      sinkhorn_kernel<<<1, 1024, 0, stream>>>(Kker, costm, cnt + (size_t)(l - 1) * KK,
                                              cnt + (size_t)l * KK, total);
    }
  }
  write_total_kernel<<<1, 1, 0, stream>>>(total, out_total);
}
